// NasRNN_7275674600340
// MI455X (gfx1250) — matrix operand order repaired and checked by reading, not compiled, not run on hardware
//
#include <hip/hip_runtime.h>

// ---------------------------------------------------------------------------
// NasRNN on MI455X (gfx1250): persistent-kernel recurrence, bf16 WMMA GEMMs.
//
// Per step: S[b, g, h] = x_t[b,:] @ W_ih[g] + m[b,:] @ W_hh[g]   (g = 0..7)
// => one fused GEMM [256 x 512] x [512 x 2048], tiled across 16 persistent
// workgroups (64x64 output tile each, all 8 gate groups local to the WG).
//
//  * Weight (B) fragment loads: single per-lane base pointer + compile-time
//    immediate offsets.
//  * One-shot f32->bf16 conversion of the whole input tensor (if ws_size
//    permits); per-step staging becomes a pure copy.
//  * global->LDS staging uses GLOBAL_LOAD_ASYNC_TO_LDS_B128 (ASYNCcnt) when
//    the builtin is available; sync fallback otherwise.
// ---------------------------------------------------------------------------

#define SEQn 1000
#define NB   256   // batch
#define NH   256   // hidden (== input)
#define NWG  16    // persistent workgroups (4x4 tiles of 64x64)
#define TB   64
#define THW  64
#define XSTR 520   // LDS X-tile row stride in bf16 elems (padded vs 512)

typedef __attribute__((ext_vector_type(16))) __bf16         v16bf;
typedef __attribute__((ext_vector_type(8)))  float          v8f;
typedef __attribute__((ext_vector_type(4)))  float          v4f;
typedef __attribute__((ext_vector_type(4)))  int            v4i;
typedef __attribute__((ext_vector_type(16))) unsigned short u16x16;
typedef __attribute__((ext_vector_type(8)))  unsigned short u16x8;

// ws layout: cnt | m double-buffer | packed weights (2 MB) | bf16 inputs
#define WS_CNT    0
#define WS_M0     256
#define WS_M1     (256 + NB * NH * 2)
#define WS_WPACK  (512 * 1024)
#define WS_XBF    (4 * 1024 * 1024)
#define XBF_BYTES ((size_t)SEQn * NB * NH * 2)   // 131,072,000 B

#if defined(__has_builtin)
#if __has_builtin(__builtin_amdgcn_global_load_async_to_lds_b128) && \
    __has_builtin(__builtin_amdgcn_s_wait_asynccnt)
#define HAVE_ASYNC_LDS 1
#endif
#endif

#ifdef HAVE_ASYNC_LDS
typedef __attribute__((address_space(1))) v4i as1_v4i;
typedef __attribute__((address_space(3))) v4i as3_v4i;
__device__ __forceinline__ as1_v4i* gcast(const void* p) {
  // flat global address == addrspace(1) address numerically
  return (as1_v4i*)(unsigned long long)(uintptr_t)p;
}
__device__ __forceinline__ as3_v4i* lcast(const void* p) {
  // ISA 10.2: for LDS-aperture flat addresses, addr[31:0] is the LDS offset
  return (as3_v4i*)(unsigned)(uintptr_t)p;
}
#endif

__device__ __forceinline__ unsigned short bfbits(float f) {
  return __builtin_bit_cast(unsigned short, (__bf16)f);
}
__device__ __forceinline__ float sigm(float x) {
  return __builtin_amdgcn_rcpf(1.0f + __expf(-x));
}
__device__ __forceinline__ float tanh_fast(float x) {
  // saturates correctly: x->+inf => 1, x->-inf => -1 (no inf/inf NaN)
  return 1.0f - 2.0f * __builtin_amdgcn_rcpf(__expf(2.0f * x) + 1.0f);
}

// ---------------------------------------------------------------------------
// Prologue 1: zero the barrier counter and the initial m buffer (bf16 zeros).
__global__ void nasrnn_init(unsigned* cnt, unsigned short* m0) {
  int tid = blockIdx.x * blockDim.x + threadIdx.x;   // 65536 threads
  if (tid < NB * NH) m0[tid] = 0;
  if (tid == 0) *cnt = 0;
}

// ---------------------------------------------------------------------------
// Prologue 2: repack [W_ih ; W_hh] (f32) into bf16 B-fragments.
// B fragment (this code's K order; A matches) for v_wmma_f32_16x16x32_bf16 (32xK x 16xN):
//   lane l: column n = l&15, K half = (l>>4)*16, 16 consecutive K values.
// Packed offset: (((g*16 + tn)*16 + kt)*32 + lane) * 16 bf16 elements.
__global__ void nasrnn_pack(const float* __restrict__ wih,
                            const float* __restrict__ whh,
                            unsigned short* __restrict__ wp) {
  int tid  = blockIdx.x * blockDim.x + threadIdx.x;  // 65536 threads
  int lane = tid & 31;
  int kt   = (tid >> 5) & 15;
  int tn   = (tid >> 9) & 15;
  int g    = tid >> 13;

  int col   = tn * 16 + (lane & 15);
  int kbase = kt * 32 + (lane >> 4) * 16;

  u16x16 o;
#pragma unroll
  for (int j = 0; j < 16; ++j) {
    int k = kbase + j;
    float w = (k < NH) ? wih[(size_t)g * NH * NH + (size_t)k * NH + col]
                       : whh[(size_t)g * NH * NH + (size_t)(k - NH) * NH + col];
    o[j] = bfbits(w);
  }
  *(u16x16*)(wp + (size_t)tid * 16) = o;
}

// ---------------------------------------------------------------------------
// Prologue 3 (optional): one-shot f32 -> bf16 conversion of all inputs.
__global__ void nasrnn_cvt(const float* __restrict__ x,
                           unsigned short* __restrict__ xb) {
  size_t i = ((size_t)blockIdx.x * 256 + threadIdx.x) * 8;   // 65,536,000 total
  v4f f0 = *(const v4f*)(x + i);
  v4f f1 = *(const v4f*)(x + i + 4);
  u16x8 h;
#pragma unroll
  for (int q = 0; q < 4; ++q) {
    h[q]     = bfbits(f0[q]);
    h[4 + q] = bfbits(f1[q]);
  }
  *(u16x8*)(xb + i) = h;
}

// ---------------------------------------------------------------------------
// Main persistent kernel: 16 blocks x 256 threads (8 waves).
template <bool XBF>
__global__ __launch_bounds__(256, 1)
void nasrnn_main(const float* __restrict__ inp,
                 const unsigned short* __restrict__ xbf,
                 const unsigned short* __restrict__ wpack,
                 unsigned short* __restrict__ mbuf0,
                 unsigned short* __restrict__ mbuf1,
                 unsigned* __restrict__ cnt,
                 float* __restrict__ out) {
  extern __shared__ char smem[];
  unsigned short* xs    = (unsigned short*)smem;             // [64][XSTR] bf16
  float*          lds_s = (float*)(smem + TB * XSTR * 2);    // [8][64*64] f32

  const int tid  = threadIdx.x;
  const int lane = tid & 31;
  const int g    = tid >> 5;                       // wave id == gate group
  const int b0   = (blockIdx.x >> 2) * TB;
  const int h0   = (blockIdx.x & 3) * THW;
  const int mrow = lane & 15;
  const int half = lane >> 4;

  // staging role: thread covers X row (tid>>2), 64 K-columns per half
  const int srow = tid >> 2;
  const int scol = (tid & 3) * 64;

  // per-lane weight fragment base; all loads use immediate offsets from here
  const unsigned short* wbase =
      wpack + (size_t)(g * 16 + (h0 >> 4)) * 512 + (size_t)lane * 16;

  // persistent per-thread state: 16 (b,h) elements of c and m
  float c_reg[16], m_reg[16];
#pragma unroll
  for (int e = 0; e < 16; ++e) { c_reg[e] = 0.f; m_reg[e] = 0.f; }
  const int ebase = tid * 16;                      // linear index in 64x64 tile
  const int erow  = ebase >> 6;
  const int ecol  = ebase & 63;

  unsigned short* mb[2] = { mbuf0, mbuf1 };
  const v8f vzero = { 0.f, 0.f, 0.f, 0.f, 0.f, 0.f, 0.f, 0.f };

  for (int t = 0; t < SEQn; ++t) {
    const unsigned short* m_prev = mb[t & 1];
    unsigned short*       m_next = mb[(t + 1) & 1];

    // ---- stage X = [x_t | m] tile into LDS (bf16) -------------------------
    {
      unsigned short* dst = xs + srow * XSTR + scol;
      if constexpr (XBF) {
        const unsigned short* src =
            xbf + (size_t)t * (NB * NH) + (size_t)(b0 + srow) * NH + scol;
#ifdef HAVE_ASYNC_LDS
#pragma unroll
        for (int j = 0; j < 64; j += 8)
          __builtin_amdgcn_global_load_async_to_lds_b128(gcast(src + j),
                                                         lcast(dst + j), 0, 0);
#else
#pragma unroll
        for (int j = 0; j < 64; j += 8)
          *(u16x8*)(dst + j) = *(const u16x8*)(src + j);
#endif
        if (t + 1 < SEQn) __builtin_prefetch(src + NB * NH, 0, 1);
      } else {
        const float* src = inp + (size_t)t * (NB * NH) +
                           (size_t)(b0 + srow) * NH + scol;
#pragma unroll
        for (int j = 0; j < 64; j += 8) {
          v4f f0 = *(const v4f*)(src + j);
          v4f f1 = *(const v4f*)(src + j + 4);
          u16x8 h;
#pragma unroll
          for (int q = 0; q < 4; ++q) {
            h[q]     = bfbits(f0[q]);
            h[4 + q] = bfbits(f1[q]);
          }
          *(u16x8*)(dst + j) = h;
        }
        if (t + 1 < SEQn) {
          __builtin_prefetch(src + NB * NH, 0, 1);
          __builtin_prefetch(src + NB * NH + 32, 0, 1);
        }
      }

      // m half: K 256..511
      const unsigned short* ms = m_prev + (size_t)(b0 + srow) * NH + scol;
      unsigned short*       md = xs + srow * XSTR + NH + scol;
#ifdef HAVE_ASYNC_LDS
#pragma unroll
      for (int j = 0; j < 64; j += 8)
        __builtin_amdgcn_global_load_async_to_lds_b128(gcast(ms + j),
                                                       lcast(md + j), 0, 0);
      __builtin_amdgcn_s_wait_asynccnt(0);
#else
#pragma unroll
      for (int j = 0; j < 64; j += 8)
        *(u16x8*)(md + j) = *(const u16x8*)(ms + j);
#endif
    }
    __syncthreads();

    // ---- GEMM: 64x64 per wave (group g), K = 512 --------------------------
    v8f acc[4][4];
#pragma unroll
    for (int r = 0; r < 4; ++r)
#pragma unroll
      for (int n = 0; n < 4; ++n) acc[r][n] = vzero;

#pragma unroll 2
    for (int kt = 0; kt < 16; ++kt) {
      v16bf afrag[4];
#pragma unroll
      for (int r = 0; r < 4; ++r) {
        // A layout: lane holds row (l&15); K halves split by l>>4.
        const unsigned short* rp =
            xs + (r * 16 + mrow) * XSTR + kt * 32 + half * 16;
        u16x8 lo = *(const u16x8*)(rp);
        u16x8 hi = *(const u16x8*)(rp + 8);
        u16x16 u = __builtin_shufflevector(lo, hi, 0, 1, 2, 3, 4, 5, 6, 7, 8,
                                           9, 10, 11, 12, 13, 14, 15);
        afrag[r] = __builtin_bit_cast(v16bf, u);
      }
#pragma unroll
      for (int n = 0; n < 4; ++n) {
        // constant offset from per-lane base => immediate-offset b128 loads
        u16x16 bw = *(const u16x16*)(wbase + ((n * 16 + kt) << 9));
        v16bf bfrag = __builtin_bit_cast(v16bf, bw);
#pragma unroll
        for (int r = 0; r < 4; ++r)
          acc[r][n] = __builtin_amdgcn_wmma_f32_16x16x32_bf16(
              false, afrag[r], false, bfrag, (short)0, acc[r][n], false, false);
      }
    }
    __syncthreads();   // X-tile fully consumed before s overwrites LDS reads

    // ---- stage s[g] tile into LDS (C/D layout: VGPR e -> M = e + 8*half) --
#pragma unroll
    for (int r = 0; r < 4; ++r)
#pragma unroll
      for (int n = 0; n < 4; ++n) {
        const int col = n * 16 + mrow;
        const int rb  = r * 16 + half * 8;
#pragma unroll
        for (int e = 0; e < 8; ++e)
          lds_s[g * 4096 + (rb + e) * 64 + col] = acc[r][n][e];
      }
    __syncthreads();

    // ---- gate network (all 8 groups local), update c/m --------------------
    unsigned short mout[16];
#pragma unroll 4
    for (int e = 0; e < 16; ++e) {
      const int idx = ebase + e;
      float s0 = lds_s[0 * 4096 + idx];
      float s1 = lds_s[1 * 4096 + idx];
      float s2 = lds_s[2 * 4096 + idx];
      float s3 = lds_s[3 * 4096 + idx];
      float s4 = lds_s[4 * 4096 + idx];
      float s5 = lds_s[5 * 4096 + idx];
      float s6 = lds_s[6 * 4096 + idx];
      float s7 = lds_s[7 * 4096 + idx];

      float l1_0 = sigm(s0);
      float l1_1 = fmaxf(s1, 0.f);
      float l1_2 = sigm(s2);
      float l1_3 = fmaxf(s3, 0.f);
      float l1_4 = tanh_fast(s4);
      float l1_5 = sigm(s5);
      float l1_6 = tanh_fast(s6);
      float l1_7 = sigm(s7);

      float l2_0 = tanh_fast(l1_0 * l1_1);
      float l2_1 = tanh_fast(l1_2 + l1_3);
      float l2_2 = tanh_fast(l1_4 * l1_5);
      float l2_3 = sigm(l1_6 + l1_7);

      float l2_0v2 = tanh_fast(l2_0 + c_reg[e]);
      float nc     = l2_0v2 * l2_1;
      float l3_1   = tanh_fast(l2_2 + l2_3);
      float nm     = tanh_fast(nc * l3_1);

      c_reg[e] = nc;
      m_reg[e] = nm;
      mout[e]  = bfbits(nm);
    }

    // ---- publish new m (16 consecutive bf16 per thread) -------------------
    {
      unsigned short* mp = m_next + (size_t)(b0 + erow) * NH + h0 + ecol;
      u16x8 lo, hi;
#pragma unroll
      for (int j = 0; j < 8; ++j) { lo[j] = mout[j]; hi[j] = mout[8 + j]; }
      *(u16x8*)(mp)     = lo;
      *(u16x8*)(mp + 8) = hi;
    }

    // ---- device-wide barrier: all 16 WGs finished step t ------------------
    __threadfence();
    __syncthreads();
    if (tid == 0) {
      __hip_atomic_fetch_add(cnt, 1u, __ATOMIC_ACQ_REL, __HIP_MEMORY_SCOPE_AGENT);
      const unsigned target = (unsigned)(t + 1) * NWG;
      while (__hip_atomic_load(cnt, __ATOMIC_ACQUIRE, __HIP_MEMORY_SCOPE_AGENT) <
             target)
        __builtin_amdgcn_s_sleep(1);
    }
    __syncthreads();
  }

  // ---- final output: c + m straight from registers ------------------------
  float* op = out + (size_t)(b0 + erow) * NH + h0 + ecol;
#pragma unroll
  for (int e = 0; e < 16; ++e) op[e] = c_reg[e] + m_reg[e];
}

// ---------------------------------------------------------------------------
extern "C" void kernel_launch(void* const* d_in, const int* in_sizes, int n_in,
                              void* d_out, int out_size, void* d_ws,
                              size_t ws_size, hipStream_t stream) {
  (void)in_sizes; (void)n_in; (void)out_size;

  const float* inp = (const float*)d_in[0];
  const float* wih = (const float*)d_in[1];
  const float* whh = (const float*)d_in[2];

  char* ws           = (char*)d_ws;
  unsigned* cnt      = (unsigned*)(ws + WS_CNT);
  unsigned short* m0 = (unsigned short*)(ws + WS_M0);
  unsigned short* m1 = (unsigned short*)(ws + WS_M1);
  unsigned short* wp = (unsigned short*)(ws + WS_WPACK);
  unsigned short* xb = (unsigned short*)(ws + WS_XBF);

  const bool use_xbf = ws_size >= (size_t)WS_XBF + XBF_BYTES;

  nasrnn_init<<<256, 256, 0, stream>>>(cnt, m0);
  nasrnn_pack<<<256, 256, 0, stream>>>(wih, whh, wp);

  const size_t lds_bytes = (size_t)TB * XSTR * 2 + 8 * 4096 * sizeof(float);
  if (use_xbf) {
    nasrnn_cvt<<<32000, 256, 0, stream>>>(inp, xb);   // 65,536,000 elems
    nasrnn_main<true><<<NWG, 256, lds_bytes, stream>>>(inp, xb, wp, m0, m1,
                                                       cnt, (float*)d_out);
  } else {
    nasrnn_main<false><<<NWG, 256, lds_bytes, stream>>>(inp, xb, wp, m0, m1,
                                                        cnt, (float*)d_out);
  }
}
